// MoE_9517647528208
// MI455X (gfx1250) — matrix operand order repaired and checked by reading, not compiled, not run on hardware
//
#include <hip/hip_runtime.h>
#include <hip/hip_bf16.h>
#include <math.h>

// ---- model config ----
#define D_DIM   1024
#define H_DIM   2048
#define E_NUM   8
#define HS_DIM  4096
#define N_TOK   8192
#define N_PAIR  (2 * N_TOK)

typedef __attribute__((ext_vector_type(16))) __bf16 v16bf;
typedef __attribute__((ext_vector_type(8)))  __bf16 v8bf;
typedef __attribute__((ext_vector_type(8)))  float  v8f;
typedef __attribute__((ext_vector_type(4)))  int    v4i;

// ---- CDNA5 async global->LDS copy (falls back to load+ds_store) ----
#if defined(__gfx1250__) && __has_builtin(__builtin_amdgcn_global_load_async_to_lds_b128)
#define HAS_ASYNC_LDS 1
typedef __attribute__((address_space(1))) v4i gas_v4i;   // global (device) AS
typedef __attribute__((address_space(3))) v4i las_v4i;   // LDS AS
#else
#define HAS_ASYNC_LDS 0
#endif

__device__ __forceinline__ void cp16(const __bf16* g, __bf16* l) {
#if HAS_ASYNC_LDS
    __builtin_amdgcn_global_load_async_to_lds_b128(
        (gas_v4i*)(v4i*)(void*)g,
        (las_v4i*)(v4i*)(void*)l, 0, 0);
#else
    *(v8bf*)l = *(const v8bf*)g;
#endif
}
__device__ __forceinline__ void cp_wait() {
#if HAS_ASYNC_LDS
    asm volatile("s_wait_asynccnt 0x0" ::: "memory");
#endif
}

// ---------------- fp32 -> bf16 conversion ----------------
__global__ __launch_bounds__(256) void cvt_kernel(const float* __restrict__ s,
                                                  __bf16* __restrict__ d, int n) {
    int i = (blockIdx.x * 256 + threadIdx.x) * 4;
    if (i < n) {
        float4 f = *(const float4*)(s + i);
        d[i + 0] = (__bf16)f.x; d[i + 1] = (__bf16)f.y;
        d[i + 2] = (__bf16)f.z; d[i + 3] = (__bf16)f.w;
    }
}

// ---------------- meta init ----------------
__global__ void init_meta(int* cnt, int* sseg) {
    int t = threadIdx.x;
    if (t < E_NUM) cnt[t] = 0;
    if (t == E_NUM) { sseg[0] = 0; sseg[1] = N_TOK; }
}

// ---------------- gate: softmax + top-2 + rank ----------------
__global__ __launch_bounds__(256) void gate_kernel(const float* __restrict__ x,
                                                   const float* __restrict__ gw,
                                                   const float* __restrict__ gb,
                                                   float* __restrict__ topw,
                                                   int* __restrict__ pair_e,
                                                   int* __restrict__ pair_rank,
                                                   int* __restrict__ cnt) {
    const int n   = blockIdx.x;
    const int tid = threadIdx.x;
    __shared__ float red[E_NUM][256];
    float acc[E_NUM];
#pragma unroll
    for (int e = 0; e < E_NUM; ++e) acc[e] = 0.0f;
    const float* xr = x + (long)n * D_DIM;
    for (int d = tid; d < D_DIM; d += 256) {
        float xv = xr[d];
#pragma unroll
        for (int e = 0; e < E_NUM; ++e) acc[e] += xv * gw[e * D_DIM + d];
    }
#pragma unroll
    for (int e = 0; e < E_NUM; ++e) red[e][tid] = acc[e];
    __syncthreads();
    for (int s = 128; s > 0; s >>= 1) {
        if (tid < s) {
#pragma unroll
            for (int e = 0; e < E_NUM; ++e) red[e][tid] += red[e][tid + s];
        }
        __syncthreads();
    }
    if (tid == 0) {
        float lg[E_NUM], p[E_NUM];
        float mx = -1e30f;
#pragma unroll
        for (int e = 0; e < E_NUM; ++e) { lg[e] = red[e][0] + gb[e]; mx = fmaxf(mx, lg[e]); }
        float sum = 0.0f;
#pragma unroll
        for (int e = 0; e < E_NUM; ++e) { p[e] = __expf(lg[e] - mx); sum += p[e]; }
        int i0 = 0;
#pragma unroll
        for (int e = 1; e < E_NUM; ++e) if (p[e] > p[i0]) i0 = e;
        int i1 = (i0 == 0) ? 1 : 0;
#pragma unroll
        for (int e = 0; e < E_NUM; ++e) if (e != i0 && p[e] > p[i1]) i1 = e;
        float w0 = p[i0] / sum, w1 = p[i1] / sum;
        float s2 = w0 + w1 + 1e-20f;
        w0 /= s2; w1 /= s2;
        topw[2 * n]     = w0;       topw[2 * n + 1]     = w1;
        pair_e[2 * n]   = i0;       pair_e[2 * n + 1]   = i1;
        pair_rank[2 * n]     = atomicAdd(&cnt[i0], 1);
        pair_rank[2 * n + 1] = atomicAdd(&cnt[i1], 1);
    }
}

// ---------------- exclusive scan over 8 expert counts ----------------
__global__ void scan_kernel(const int* __restrict__ cnt, int* __restrict__ off) {
    if (threadIdx.x == 0 && blockIdx.x == 0) {
        int s = 0;
        for (int e = 0; e < E_NUM; ++e) { off[e] = s; s += cnt[e]; }
        off[E_NUM] = s;
    }
}

// ---------------- compact (token,slot) <-> position maps ----------------
__global__ __launch_bounds__(256) void build_pairs(const int* __restrict__ pe,
                                                   const int* __restrict__ pr,
                                                   const int* __restrict__ off,
                                                   int* __restrict__ ppos,
                                                   int* __restrict__ ptok) {
    int id = blockIdx.x * 256 + threadIdx.x;
    if (id < N_PAIR) {
        int p = off[pe[id]] + pr[id];
        ppos[id] = p;
        ptok[p] = id >> 1;
    }
}

// ---------------- WMMA bf16 GEMM (optionally fused dual-B SwiGLU) ----------------
// Block tile 128x64, BK=32, 8 waves of 32x32 (4x2), double-buffered LDS,
// async global->LDS staging, v_wmma_f32_16x16x32_bf16.
template <int DUAL>
__global__ __launch_bounds__(256) void gemm_ffn(const __bf16* __restrict__ A,
                                                const __bf16* __restrict__ B0,
                                                const __bf16* __restrict__ B1,
                                                const float* __restrict__ bias0,
                                                const float* __restrict__ bias1,
                                                long wStride, int biasStride,
                                                const int* __restrict__ seg_off,
                                                const int* __restrict__ seg_cnt,
                                                const int* __restrict__ row_idx,
                                                void* __restrict__ outP,
                                                int ldOut, int K) {
    const int e   = blockIdx.z;
    const int cnt = seg_cnt[e];
    const int mBase = blockIdx.y * 128;
    if (mBase >= cnt) return;                 // early-exit: empty tile for this expert
    const int off   = seg_off[e];
    const int nBase = blockIdx.x * 64;

    const __bf16* Bp0 = B0 + (long)e * wStride;
    const __bf16* Bp1 = DUAL ? (B1 + (long)e * wStride) : B0;

    constexpr int LDT = 40;                   // padded K stride (elements)
    constexpr int ASZ = 128 * LDT;            // one A stage
    constexpr int BSZ = 64 * LDT;             // one B stage
    __shared__ __bf16 As [2 * ASZ];
    __shared__ __bf16 Bs0[2 * BSZ];
    __shared__ __bf16 Bs1[DUAL ? 2 * BSZ : 8];

    const int tid   = threadIdx.x;
    const int wid   = tid >> 5;
    const int lane  = tid & 31;
    const int wm    = (wid >> 1) * 32;        // 4x2 wave grid -> 128x64 block tile
    const int wn    = (wid & 1)  * 32;
    const int lhalf = lane >> 4;              // 0/1
    const int l16   = lane & 15;

    // ---- K-invariant staging setup (gather hoisted out of the K loop) ----
    int ar[2], ac[2];
    const __bf16* aSrc[2];
#pragma unroll
    for (int ch = 0; ch < 2; ++ch) {
        int idx = ch * 256 + tid;             // 512 chunks of 8 bf16 (128x32)
        ar[ch] = idx >> 2;
        ac[ch] = (idx & 3) * 8;
        int m = mBase + ar[ch];
        if (m >= cnt) m = cnt - 1;            // clamp: finite data, rows masked on store
        int g   = off + m;
        int src = row_idx ? row_idx[g] : g;
        aSrc[ch] = A + (long)src * K + ac[ch];
    }
    const int br = tid >> 2;                  // 256 chunks (64x32) per B matrix
    const int bc = (tid & 3) * 8;
    const __bf16* bSrc0 = Bp0 + (long)(nBase + br) * K + bc;
    const __bf16* bSrc1 = Bp1 + (long)(nBase + br) * K + bc;

    auto stage = [&](int buf, int kb) {
#pragma unroll
        for (int ch = 0; ch < 2; ++ch)
            cp16(aSrc[ch] + kb, As + buf * ASZ + ar[ch] * LDT + ac[ch]);
        cp16(bSrc0 + kb, Bs0 + buf * BSZ + br * LDT + bc);
        if (DUAL) cp16(bSrc1 + kb, Bs1 + buf * BSZ + br * LDT + bc);
    };

    const v8f zero = {0.f, 0.f, 0.f, 0.f, 0.f, 0.f, 0.f, 0.f};
    v8f acc0[2][2], acc1[2][2];
#pragma unroll
    for (int mi = 0; mi < 2; ++mi)
#pragma unroll
        for (int ni = 0; ni < 2; ++ni) { acc0[mi][ni] = zero; acc1[mi][ni] = zero; }

    stage(0, 0);                              // prologue

    for (int kb = 0; kb < K; kb += 32) {
        const int cur = (kb >> 5) & 1;
        cp_wait();                            // own async chunks for buffer `cur` done
        __syncthreads();                      // everyone's chunks visible
        if (kb + 32 < K) stage(1 - cur, kb + 32);

        const __bf16* Ac  = As  + cur * ASZ;
        const __bf16* Bc0 = Bs0 + cur * BSZ;
        const __bf16* Bc1 = Bs1 + (DUAL ? cur * BSZ : 0);

        // ---- fragments (ISA 7.12.2 layouts) ----
        v16bf af[2], bf0[2], bf1[2];
#pragma unroll
        for (int mi = 0; mi < 2; ++mi) {
            int row = wm + mi * 16 + l16;
            int k0  = lhalf * 8;
            v8bf lo = *(const v8bf*)(Ac + row * LDT + k0);
            v8bf hi = *(const v8bf*)(Ac + row * LDT + 16 + k0);
#pragma unroll
            for (int i = 0; i < 8; ++i) { af[mi][i] = lo[i]; af[mi][8 + i] = hi[i]; }
        }
#pragma unroll
        for (int ni = 0; ni < 2; ++ni) {
            int row = wn + ni * 16 + l16;
            int k0  = lhalf * 8;
            v8bf lo = *(const v8bf*)(Bc0 + row * LDT + k0);
            v8bf hi = *(const v8bf*)(Bc0 + row * LDT + 16 + k0);
#pragma unroll
            for (int i = 0; i < 8; ++i) { bf0[ni][i] = lo[i]; bf0[ni][8 + i] = hi[i]; }
            if (DUAL) {
                v8bf lo1 = *(const v8bf*)(Bc1 + row * LDT + k0);
                v8bf hi1 = *(const v8bf*)(Bc1 + row * LDT + 16 + k0);
#pragma unroll
                for (int i = 0; i < 8; ++i) { bf1[ni][i] = lo1[i]; bf1[ni][8 + i] = hi1[i]; }
            }
        }
        // ---- WMMA ----
#pragma unroll
        for (int mi = 0; mi < 2; ++mi)
#pragma unroll
            for (int ni = 0; ni < 2; ++ni) {
                acc0[mi][ni] = __builtin_amdgcn_wmma_f32_16x16x32_bf16(
                    false, af[mi], false, bf0[ni], (short)0, acc0[mi][ni], false, false);
                if (DUAL)
                    acc1[mi][ni] = __builtin_amdgcn_wmma_f32_16x16x32_bf16(
                        false, af[mi], false, bf1[ni], (short)0, acc1[mi][ni], false, false);
            }
        __syncthreads();                      // protect `cur` from restaging
    }

    // ---- epilogue: C layout lane->(M = v + 8*lhalf, N = l16) ----
#pragma unroll
    for (int mi = 0; mi < 2; ++mi)
#pragma unroll
        for (int ni = 0; ni < 2; ++ni) {
            const int col = nBase + wn + ni * 16 + l16;
#pragma unroll
            for (int v = 0; v < 8; ++v) {
                int m = mBase + wm + mi * 16 + lhalf * 8 + v;
                if (m < cnt) {
                    long g = off + m;
                    if (DUAL) {
                        float a = acc0[mi][ni][v] + bias0[(long)e * biasStride + col];
                        float b = acc1[mi][ni][v] + bias1[(long)e * biasStride + col];
                        float s = a / (1.0f + __expf(-a));        // silu(a) * b
                        ((__bf16*)outP)[g * ldOut + col] = (__bf16)(s * b);
                    } else {
                        ((float*)outP)[g * ldOut + col] =
                            acc0[mi][ni][v] + bias0[(long)e * biasStride + col];
                    }
                }
            }
        }
}

// ---------------- combine routed slots into output ----------------
__global__ __launch_bounds__(256) void combine_kernel(const float* __restrict__ y2,
                                                      const float* __restrict__ topw,
                                                      const int* __restrict__ ppos,
                                                      float* __restrict__ out) {
    int n  = blockIdx.x;
    int p0 = ppos[2 * n], p1 = ppos[2 * n + 1];
    float w0 = topw[2 * n], w1 = topw[2 * n + 1];
    const float* r0 = y2 + (long)p0 * D_DIM;
    const float* r1 = y2 + (long)p1 * D_DIM;
    float* o = out + (long)n * D_DIM;
    for (int d = threadIdx.x; d < D_DIM; d += 256)
        o[d] += w0 * r0[d] + w1 * r1[d];
}

// ---------------- host ----------------
static inline void launch_cvt(const float* s, __bf16* d, long n, hipStream_t st) {
    int blocks = (int)((n / 4 + 255) / 256);
    cvt_kernel<<<blocks, 256, 0, st>>>(s, d, (int)n);
}

extern "C" void kernel_launch(void* const* d_in, const int* in_sizes, int n_in,
                              void* d_out, int out_size, void* d_ws, size_t ws_size,
                              hipStream_t stream) {
    (void)in_sizes; (void)n_in; (void)out_size; (void)ws_size;
    const float* x   = (const float*)d_in[0];
    const float* gw  = (const float*)d_in[1];
    const float* gb  = (const float*)d_in[2];
    const float* w1  = (const float*)d_in[3];
    const float* b1  = (const float*)d_in[4];
    const float* w2  = (const float*)d_in[5];
    const float* b2  = (const float*)d_in[6];
    const float* w3  = (const float*)d_in[7];
    const float* b3  = (const float*)d_in[8];
    const float* sw1 = (const float*)d_in[9];
    const float* sb1 = (const float*)d_in[10];
    const float* sw2 = (const float*)d_in[11];
    const float* sb2 = (const float*)d_in[12];
    const float* sw3 = (const float*)d_in[13];
    const float* sb3 = (const float*)d_in[14];
    float* out = (float*)d_out;

    char* ws = (char*)d_ws;
    size_t o = 0;
    auto alloc = [&](size_t bytes) -> char* {
        o = (o + 255) & ~(size_t)255;
        char* p = ws + o; o += bytes; return p;
    };

    __bf16* xb   = (__bf16*)alloc((size_t)N_TOK * D_DIM * 2);
    __bf16* w1b  = (__bf16*)alloc((size_t)E_NUM * H_DIM * D_DIM * 2);
    __bf16* w3b  = (__bf16*)alloc((size_t)E_NUM * H_DIM * D_DIM * 2);
    __bf16* w2b  = (__bf16*)alloc((size_t)E_NUM * D_DIM * H_DIM * 2);
    __bf16* sw1b = (__bf16*)alloc((size_t)HS_DIM * D_DIM * 2);
    __bf16* sw3b = (__bf16*)alloc((size_t)HS_DIM * D_DIM * 2);
    __bf16* sw2b = (__bf16*)alloc((size_t)D_DIM * HS_DIM * 2);
    __bf16* hid  = (__bf16*)alloc((size_t)N_PAIR * H_DIM * 2);
    __bf16* shid = (__bf16*)alloc((size_t)N_TOK * HS_DIM * 2);
    float*  y2   = (float*)alloc((size_t)N_PAIR * D_DIM * 4);
    float*  topw = (float*)alloc((size_t)N_PAIR * 4);
    int* pair_e    = (int*)alloc((size_t)N_PAIR * 4);
    int* pair_rank = (int*)alloc((size_t)N_PAIR * 4);
    int* pair_pos  = (int*)alloc((size_t)N_PAIR * 4);
    int* pair_tok  = (int*)alloc((size_t)N_PAIR * 4);
    int* cnt = (int*)alloc(E_NUM * 4);
    int* off = (int*)alloc((E_NUM + 1) * 4);
    int* sseg = (int*)alloc(2 * 4);

    // 1) meta init + conversions
    init_meta<<<1, 32, 0, stream>>>(cnt, sseg);
    launch_cvt(x,   xb,   (long)N_TOK * D_DIM, stream);
    launch_cvt(w1,  w1b,  (long)E_NUM * H_DIM * D_DIM, stream);
    launch_cvt(w3,  w3b,  (long)E_NUM * H_DIM * D_DIM, stream);
    launch_cvt(w2,  w2b,  (long)E_NUM * D_DIM * H_DIM, stream);
    launch_cvt(sw1, sw1b, (long)HS_DIM * D_DIM, stream);
    launch_cvt(sw3, sw3b, (long)HS_DIM * D_DIM, stream);
    launch_cvt(sw2, sw2b, (long)D_DIM * HS_DIM, stream);

    // 2) gating + dispatch structures
    gate_kernel<<<N_TOK, 256, 0, stream>>>(x, gw, gb, topw, pair_e, pair_rank, cnt);
    scan_kernel<<<1, 1, 0, stream>>>(cnt, off);
    build_pairs<<<N_PAIR / 256, 256, 0, stream>>>(pair_e, pair_rank, off, pair_pos, pair_tok);

    // 3) routed experts: fused SwiGLU GEMM (X -> hidden), then hidden -> y2
    gemm_ffn<1><<<dim3(H_DIM / 64, N_TOK / 128, E_NUM), 256, 0, stream>>>(
        xb, w1b, w3b, b1, b3, (long)H_DIM * D_DIM, H_DIM,
        off, cnt, pair_tok, hid, H_DIM, D_DIM);
    gemm_ffn<0><<<dim3(D_DIM / 64, N_TOK / 128, E_NUM), 256, 0, stream>>>(
        hid, w2b, nullptr, b2, nullptr, (long)D_DIM * H_DIM, D_DIM,
        off, cnt, nullptr, y2, D_DIM, H_DIM);

    // 4) shared expert: dense SwiGLU, second GEMM writes d_out directly
    gemm_ffn<1><<<dim3(HS_DIM / 64, N_TOK / 128, 1), 256, 0, stream>>>(
        xb, sw1b, sw3b, sb1, sb3, 0L, 0,
        sseg, sseg + 1, nullptr, shid, HS_DIM, D_DIM);
    gemm_ffn<0><<<dim3(D_DIM / 64, N_TOK / 128, 1), 256, 0, stream>>>(
        shid, sw2b, nullptr, sb2, nullptr, 0L, 0,
        sseg, sseg + 1, nullptr, out, D_DIM, HS_DIM);

    // 5) add weighted routed contributions
    combine_kernel<<<N_TOK, 256, 0, stream>>>(y2, topw, pair_pos, out);
}
